// DistanceTokenBias_37546604101766
// MI455X (gfx1250) — matrix operand order repaired and checked by reading, not compiled, not run on hardware
//
#include <hip/hip_runtime.h>
#include <hip/hip_bf16.h>

// ---------------- types for WMMA ----------------
typedef __attribute__((ext_vector_type(16))) _Float16 v16h;
typedef __attribute__((ext_vector_type(8)))  _Float16 v8h;
typedef __attribute__((ext_vector_type(8)))  float    v8f;

// ---------------- problem constants ----------------
#define N_TOK   384
#define A_ATOM  1536
#define BMULT   4
#define ZDIM    128      // rpe / hidden dim
#define HID     128
#define ODIM    16
#define FDIM    16
#define DFOUR   256
#define INFEAT  274      // 16 + 128 + 1 + 1 + 128
#define KPAD    288      // 9 * 32 (GEMM K extent)
#define NWAVES  4
#define JT      6        // j-tiles per wave: 6 * 4 waves * 16 = 384 = N_TOK

// LDS row strides chosen for conflict-free b128 reads:
//  KSTR = 296 halves = 148 dwords; n*148 mod 64 = n*20 mod 64 -> 16 distinct
//  multiples of 4 for n = 0..15 -> disjoint 4-dword bank ranges per lane.
#define KSTR    296
#define HSTR    40       // 20 dwords -> same property

// GaussianSmearing constants: offset[g] = g * (2/127), coeff = -0.5/(2/127)^2
#define GSTEP   (2.0f / 127.0f)
#define GCOEFF  (-2016.125f)

// ---------------- LDS layout (dynamic, bytes) ----------------
#define W1_BYTES   (HID * KSTR * 2)                 // 75776
#define W3_BYTES   (ODIM * KSTR * 2)                //  9472
#define WPACK_BYTES (2 * W1_BYTES + W3_BYTES)       // 161024 (16B multiple)
#define XN_OFF     WPACK_BYTES
#define XN_BYTES   (NWAVES * 16 * KSTR * 2)         // 37888
#define H_OFF      (XN_OFF + XN_BYTES)
#define H_BYTES    (NWAVES * 16 * HSTR * 2)         //  5120
#define SMEM_BYTES (H_OFF + H_BYTES)                // 204032 < 320KB WGP LDS

// ---------------- prologue kernels ----------------
__global__ void prep_fd_kernel(const float* __restrict__ nf,
                               const float* __restrict__ Wf,
                               float* __restrict__ fd) {
    int idx = blockIdx.x * blockDim.x + threadIdx.x;   // BMULT*FDIM = 64
    if (idx >= BMULT * FDIM) return;
    int b = idx >> 4, o = idx & 15;
    float s = 0.f;
    for (int k = 0; k < DFOUR; ++k) s += nf[b * DFOUR + k] * Wf[o * DFOUR + k];
    fd[idx] = s;
}

__global__ void prep_repr_kernel(const float* __restrict__ t2ra,  // [1,N,A]
                                 const float* __restrict__ rn,    // [BM,A,3]
                                 float* __restrict__ rr) {        // [BM,N,3]
    int idx = blockIdx.x * blockDim.x + threadIdx.x;   // BM*N*3 = 4608
    if (idx >= BMULT * N_TOK * 3) return;
    int c = idx % 3;
    int n = (idx / 3) % N_TOK;
    int b = idx / (3 * N_TOK);
    const float* t = t2ra + (size_t)n * A_ATOM;
    const float* r = rn + (size_t)b * A_ATOM * 3 + c;
    float s = 0.f;
    for (int a = 0; a < A_ATOM; ++a) s += t[a] * r[a * 3];
    rr[idx] = s;
}

// Pack f16 weights, padded to KSTR-stride rows, contiguous: [W1 | W2 | W3]
__global__ void prep_weights_kernel(const float* __restrict__ W1,
                                    const float* __restrict__ W2,
                                    const float* __restrict__ W3,
                                    _Float16* __restrict__ Wpack) {
    int idx = blockIdx.x * blockDim.x + threadIdx.x;
    _Float16* W1h = Wpack;
    _Float16* W2h = Wpack + HID * KSTR;
    _Float16* W3h = Wpack + 2 * HID * KSTR;
    if (idx < HID * KSTR) {
        int n = idx / KSTR, k = idx % KSTR;
        W1h[idx] = (_Float16)(k < INFEAT ? W1[n * INFEAT + k] : 0.f);
        W2h[idx] = (_Float16)(k < INFEAT ? W2[n * INFEAT + k] : 0.f);
    }
    if (idx < ODIM * KSTR) {
        int n = idx / KSTR, k = idx % KSTR;
        W3h[idx] = (_Float16)(k < HID ? W3[n * HID + k] : 0.f);
    }
}

// ---------------- main fused kernel ----------------
__device__ __forceinline__ float wave_sum32(float v) {
    #pragma unroll
    for (int m = 16; m; m >>= 1) v += __shfl_xor(v, m, 32);
    return v;
}

union ABfrag { v16h v; v8h h[2]; };

__global__ __launch_bounds__(NWAVES * 32, 1)
void dtb_main_kernel(const float* __restrict__ fd,     // [BM,16]
                     const float* __restrict__ rr,     // [BM,N,3]
                     const float* __restrict__ rpe,    // [1,N,N,Z]
                     const float* __restrict__ lng,    // [274]
                     const float* __restrict__ lnb,    // [274]
                     const _Float16* __restrict__ Wpack, // padded [W1|W2|W3]
                     float* __restrict__ out) {        // [BM,16,N,N]
    extern __shared__ __align__(16) char smem[];
    _Float16* sW1 = (_Float16*)smem;
    _Float16* sW2 = (_Float16*)(smem + W1_BYTES);
    _Float16* sW3 = (_Float16*)(smem + 2 * W1_BYTES);
    _Float16* sXN = (_Float16*)(smem + XN_OFF);   // [NWAVES][16][KSTR]
    _Float16* sH  = (_Float16*)(smem + H_OFF);    // [NWAVES][16][HSTR]

    const int lane = threadIdx.x & 31;
    const int wave = threadIdx.x >> 5;
    const int b = blockIdx.y;
    const int i = blockIdx.x;

    // ---- stage packed weights into LDS once per block (linear b128 copies) ----
    {
        uint4* dst = (uint4*)smem;
        const uint4* src = (const uint4*)Wpack;
        for (int idx = threadIdx.x; idx < WPACK_BYTES / 16; idx += NWAVES * 32)
            dst[idx] = src[idx];
    }
    __syncthreads();

    // per-lane LN params for features f = lane + 32k
    float gam[9], bet[9];
    #pragma unroll
    for (int k = 0; k < 9; ++k) {
        int f = lane + 32 * k;
        gam[k] = (f < INFEAT) ? lng[f] : 0.f;
        bet[k] = (f < INFEAT) ? lnb[f] : 0.f;
    }
    const float fdv = (lane < 16) ? fd[b * FDIM + lane] : 0.f;
    const float ri0 = rr[((size_t)b * N_TOK + i) * 3 + 0];
    const float ri1 = rr[((size_t)b * N_TOK + i) * 3 + 1];
    const float ri2 = rr[((size_t)b * N_TOK + i) * 3 + 2];

    _Float16* xn0 = sXN + (size_t)wave * 16 * KSTR;   // this wave's xn tile
    _Float16* h0  = sH  + (size_t)wave * 16 * HSTR;   // this wave's h staging

    for (int t = 0; t < JT; ++t) {
        const int j0 = (t * NWAVES + wave) * 16;

        // safe pairwise distance (lanes 0..15 own one pair each)
        float dlane;
        {
            int j = j0 + (lane & 15);
            const float* rj = rr + ((size_t)b * N_TOK + j) * 3;
            float dx = ri0 - rj[0], dy = ri1 - rj[1], dz = ri2 - rj[2];
            float sq = dx * dx + dy * dy + dz * dz;
            dlane = sq > 0.f ? sqrtf(sq) : 0.f;
        }

        // ---- build layer-normed feature rows xn[16][KPAD] (f16) in LDS ----
        for (int p = 0; p < 16; ++p) {
            float d = __shfl(dlane, p, 32);
            const float* rp = rpe + ((size_t)i * N_TOK + (j0 + p)) * ZDIM;
            float x[9];
            float s = 0.f, ss = 0.f;
            #pragma unroll
            for (int k = 0; k < 9; ++k) {
                int f = lane + 32 * k;
                float v;
                if (f < 16)        v = fdv;                                 // fourier proj
                else if (f < 144)  { float u = d - (float)(f - 16) * GSTEP; // gaussian
                                     v = __expf(GCOEFF * u * u); }
                else if (f == 144) v = d;                                   // distance
                else if (f == 145) v = 0.f;                                 // zeros slot
                else if (f < 274)  v = rp[f - 146];                         // rpe
                else               v = 0.f;                                 // K padding
                x[k] = v; s += v; ss += v * v;
            }
            s  = wave_sum32(s);
            ss = wave_sum32(ss);
            float mu  = s * (1.f / (float)INFEAT);
            float var = ss * (1.f / (float)INFEAT) - mu * mu;
            float rs  = rsqrtf(var + 1e-5f);
            #pragma unroll
            for (int k = 0; k < 9; ++k) {
                int f = lane + 32 * k;
                float xv = (x[k] - mu) * rs * gam[k] + bet[k];
                xn0[p * KSTR + f] = (_Float16)((f < INFEAT) ? xv : 0.f);
            }
        }
        __syncthreads();

        // prefetch next tile's rpe rows into cache while we compute
        if (t + 1 < JT) {
            int jn = ((t + 1) * NWAVES + wave) * 16 + (lane & 15);
            const char* pf = (const char*)(rpe + ((size_t)i * N_TOK + jn) * ZDIM);
            __builtin_prefetch(pf + (lane >> 4) * 256, 0, 0);
        }

        // ---- A fragments in regs: 16x32 f16, m=lane&15, halves by lane>>4 ----
        ABfrag af[9];
        {
            int m = lane & 15, hb = (lane >> 4) * 8;
            #pragma unroll
            for (int kc = 0; kc < 9; ++kc) {
                const _Float16* base = xn0 + m * KSTR + kc * 32 + hb;
                af[kc].h[0] = *(const v8h*)(base);
                af[kc].h[1] = *(const v8h*)(base + 16);
            }
        }

        // B fragment loader from LDS weights: n=lane&15, K split by lane>>4
        auto loadB = [&](const _Float16* W, int nt, int kc) -> v16h {
            int n = nt * 16 + (lane & 15);
            const _Float16* base = W + (size_t)n * KSTR + kc * 32 + (lane >> 4) * 8;
            ABfrag r;
            r.h[0] = *(const v8h*)(base);
            r.h[1] = *(const v8h*)(base + 16);
            return r.v;
        };

        v8f acc3 = {0.f, 0.f, 0.f, 0.f, 0.f, 0.f, 0.f, 0.f};
        for (int q = 0; q < 4; ++q) {       // 4 K-chunks of the W3 projection
            #pragma unroll
            for (int half = 0; half < 2; ++half) {
                int nt = q * 2 + half;      // hidden tile
                v8f a1 = {0.f,0.f,0.f,0.f,0.f,0.f,0.f,0.f};
                v8f a2 = {0.f,0.f,0.f,0.f,0.f,0.f,0.f,0.f};
                #pragma unroll
                for (int kc = 0; kc < 9; ++kc) {
                    v16h b1 = loadB(sW1, nt, kc);
                    a1 = __builtin_amdgcn_wmma_f32_16x16x32_f16(
                            false, af[kc].v, false, b1, (short)0, a1, false, false);
                    v16h b2 = loadB(sW2, nt, kc);
                    a2 = __builtin_amdgcn_wmma_f32_16x16x32_f16(
                            false, af[kc].v, false, b2, (short)0, a2, false, false);
                }
                // SwiGLU: h = silu(u) * v ; stage f16 for the W3 A-fragment
                int n = lane & 15, mb = (lane >> 4) * 8;
                #pragma unroll
                for (int r = 0; r < 8; ++r) {
                    float u = a1[r], vv = a2[r];
                    float hval = (u / (1.f + __expf(-u))) * vv;
                    h0[(mb + r) * HSTR + half * 16 + n] = (_Float16)hval;
                }
            }
            __syncthreads();
            // h-chunk (16x32) @ W3 chunk -> accumulate 16x16 output
            ABfrag ha;
            {
                int m = lane & 15, hb = (lane >> 4) * 8;
                const _Float16* base = h0 + m * HSTR + hb;
                ha.h[0] = *(const v8h*)(base);
                ha.h[1] = *(const v8h*)(base + 16);
            }
            v16h b3 = loadB(sW3, 0, 0);     // placeholder, replaced below
            {
                int n = lane & 15;
                const _Float16* base = sW3 + (size_t)n * KSTR + q * 32 + (lane >> 4) * 8;
                ABfrag r2;
                r2.h[0] = *(const v8h*)(base);
                r2.h[1] = *(const v8h*)(base + 16);
                b3 = r2.v;
            }
            acc3 = __builtin_amdgcn_wmma_f32_16x16x32_f16(
                      false, ha.v, false, b3, (short)0, acc3, false, false);
            __syncthreads();
        }

        // ---- store out[b][o][i][j]; D layout: m = r + 8*(lane>>4), n = lane&15 ----
        {
            int o = lane & 15, mb = (lane >> 4) * 8;
            #pragma unroll
            for (int r = 0; r < 8; ++r) {
                int j = j0 + mb + r;
                out[(((size_t)b * ODIM + o) * N_TOK + i) * N_TOK + j] = acc3[r];
            }
        }
        __syncthreads();   // protect xn/h staging before next tile overwrites
    }
}

// ---------------- launch ----------------
extern "C" void kernel_launch(void* const* d_in, const int* in_sizes, int n_in,
                              void* d_out, int out_size, void* d_ws, size_t ws_size,
                              hipStream_t stream) {
    const float* nf   = (const float*)d_in[0];  // normed_fourier [4,256]
    const float* rn   = (const float*)d_in[1];  // r_noisy [4,1536,3]
    const float* rpe  = (const float*)d_in[2];  // rpe [1,384,384,128]
    const float* t2ra = (const float*)d_in[3];  // token_to_rep_atom [1,384,1536]
    const float* Wf   = (const float*)d_in[4];  // W_fourier [16,256]
    const float* lng  = (const float*)d_in[5];  // ln_gamma [274]
    const float* lnb  = (const float*)d_in[6];  // ln_beta [274]
    const float* W1   = (const float*)d_in[7];  // [128,274]
    const float* W2   = (const float*)d_in[8];  // [128,274]
    const float* W3   = (const float*)d_in[9];  // [16,128]
    float* out = (float*)d_out;                 // [4,16,384,384] f32

    // workspace layout (16B-aligned slices)
    char* ws = (char*)d_ws;
    float*    fd    = (float*)(ws);                     //   256 B
    float*    rr    = (float*)(ws + 256);               // 18432 B
    _Float16* Wpack = (_Float16*)(ws + 256 + 18432);    // 161024 B padded f16

    prep_fd_kernel<<<1, 64, 0, stream>>>(nf, Wf, fd);
    prep_repr_kernel<<<(BMULT * N_TOK * 3 + 255) / 256, 256, 0, stream>>>(t2ra, rn, rr);
    prep_weights_kernel<<<(HID * KSTR + 255) / 256, 256, 0, stream>>>(W1, W2, W3, Wpack);

    // allow > default dynamic LDS (204 KB of the WGP's 320 KB)
    hipFuncSetAttribute((const void*)dtb_main_kernel,
                        hipFuncAttributeMaxDynamicSharedMemorySize, SMEM_BYTES);

    dim3 grid(N_TOK, BMULT);                  // (384 rows, 4 batches)
    dtb_main_kernel<<<grid, NWAVES * 32, SMEM_BYTES, stream>>>(
        fd, rr, rpe, lng, lnb, Wpack, out);
}
